// QNet_12884901888570
// MI455X (gfx1250) — matrix operand order repaired and checked by reading, not compiled, not run on hardware
//
#include <hip/hip_runtime.h>
#include <hip/hip_bf16.h>
#include <stdint.h>

#define BATCH 4096
#define NAG 8
#define OBS 64
#define HID 64
#define ACT 16
#define GHD 128
#define G3 384

typedef __attribute__((ext_vector_type(16))) __bf16 v16bf;
typedef __attribute__((ext_vector_type(8)))  float  v8f;

union Frag16 { v16bf v; uint4 q[2]; };

__device__ __forceinline__ v8f wmma_bf(v16bf a, v16bf b, v8f c) {
  return __builtin_amdgcn_wmma_f32_16x16x32_bf16(false, a, false, b, (short)0, c, false, false);
}

// fast sigmoid: v_exp_f32 + v_rcp_f32 (saturates correctly at +/-inf)
__device__ __forceinline__ float fast_sigmoid(float x) {
  return __builtin_amdgcn_rcpf(1.f + __expf(-x));
}
// fast tanh: 1 - 2/(exp(2x)+1)
__device__ __forceinline__ float fast_tanh(float x) {
  return 1.f - 2.f * __builtin_amdgcn_rcpf(__expf(2.f * x) + 1.f);
}

// ---------------- Kernel 1: per-agent encode MLP (fp32) ----------------
__global__ __launch_bounds__(64) void encode_kernel(
    const float* __restrict__ obs, const float* __restrict__ W1, const float* __restrict__ b1,
    const float* __restrict__ W2, const float* __restrict__ b2,
    float* __restrict__ h, __bf16* __restrict__ hbf) {
  __shared__ float tl[64 * 129];
  const int n = blockIdx.y;
  const int tid = threadIdx.x;
  const int b = blockIdx.x * 64 + tid;
  const float* op = obs + ((size_t)b * NAG + n) * OBS;
  float o[OBS];
#pragma unroll
  for (int c = 0; c < OBS; ++c) o[c] = op[c];
  const float* w1 = W1 + (size_t)n * OBS * GHD;
  const float* b1p = b1 + n * GHD;
  for (int g = 0; g < GHD; ++g) {
    float acc = b1p[g];
#pragma unroll
    for (int c = 0; c < OBS; ++c) acc = fmaf(o[c], w1[c * GHD + g], acc);
    tl[tid * 129 + g] = fmaxf(acc, 0.f);
  }
  const float* w2 = W2 + (size_t)n * GHD * HID;
  const float* b2p = b2 + n * HID;
  float* hp = h + ((size_t)b * NAG + n) * HID;
  __bf16* hbp = hbf + ((size_t)b * NAG + n) * HID;
  for (int a = 0; a < HID; ++a) {
    float acc = b2p[a];
#pragma unroll 16
    for (int g = 0; g < GHD; ++g) acc = fmaf(tl[tid * 129 + g], w2[g * HID + a], acc);
    float hv = fmaxf(acc, 0.f);
    hp[a] = hv;
    hbp[a] = (__bf16)hv;   // native f32->bf16 cvt
  }
}

// ---- Kernel 2: pairwise GRU gates via WMMA + gate head + hard gumbel ----
__global__ __launch_bounds__(256) void pair_kernel(
    const __bf16* __restrict__ hbf, const float* __restrict__ gumbel,
    const float* __restrict__ Wi, const float* __restrict__ bi, const float* __restrict__ bh,
    const float* __restrict__ Wg, const float* __restrict__ bg,
    float* __restrict__ wout) {
  __shared__ __align__(16) __bf16 WiL[G3 * 136];  // [g][c] bf16, padded row stride
  __shared__ float biL[G3];
  __shared__ float bhL[G3];
  __shared__ float WgL[2 * GHD];
  __shared__ float bgL[2];
  const int pair = blockIdx.y;
  const int i = pair >> 3, j = pair & 7;
  const int tid = threadIdx.x;

  const float* WiG = Wi + (size_t)pair * (G3 * GHD);
  for (int e = tid; e < G3 * GHD; e += 256)
    WiL[(e >> 7) * 136 + (e & 127)] = (__bf16)WiG[e];   // native cvt
  for (int e = tid; e < G3; e += 256) { biL[e] = bi[pair * G3 + e]; bhL[e] = bh[pair * G3 + e]; }
  for (int e = tid; e < 2 * GHD; e += 256) WgL[e] = Wg[pair * 2 * GHD + e];
  if (tid < 2) bgL[tid] = bg[pair * 2 + tid];
  __syncthreads();

  const int wave = tid >> 5, lane = tid & 31;
  const int lhalf = lane >> 4, ncol = lane & 15;

  for (int it = 0; it < 4; ++it) {
    const int tileIdx = blockIdx.x * 32 + it * 8 + wave;
    const int mbase = tileIdx * 16;
    // A fragments: 4 K-steps of 32; steps 0,1 read h_i cols [0,32)/[32,64), steps 2,3 read h_j
    Frag16 af[4];
#pragma unroll
    for (int s = 0; s < 4; ++s) {
      const int agent = (s < 2) ? i : j;
      const int row = mbase + ncol;
      const __bf16* p =
          hbf + ((size_t)row * NAG + agent) * HID + (s & 1) * 32 + lhalf * 16;
      af[s].q[0] = *(const uint4*)p;         // k_local {base..base+7}
      af[s].q[1] = *(const uint4*)(p + 8);   // k_local {base+8..base+15}
    }
    float pl0[8], pl1[8];
#pragma unroll
    for (int e = 0; e < 8; ++e) { pl0[e] = 0.f; pl1[e] = 0.f; }

    for (int t = 0; t < 8; ++t) {
      v8f aR = {0.f,0.f,0.f,0.f,0.f,0.f,0.f,0.f};
      v8f aZ = {0.f,0.f,0.f,0.f,0.f,0.f,0.f,0.f};
      v8f aN = {0.f,0.f,0.f,0.f,0.f,0.f,0.f,0.f};
#pragma unroll
      for (int s = 0; s < 4; ++s) {
        Frag16 bfrag;
        const __bf16* p = WiL + (t * 16 + ncol) * 136 + s * 32 + lhalf * 16;
        bfrag.q[0] = *(const uint4*)p;
        bfrag.q[1] = *(const uint4*)(p + 8);
        aR = wmma_bf(af[s].v, bfrag.v, aR);
      }
#pragma unroll
      for (int s = 0; s < 4; ++s) {
        Frag16 bfrag;
        const __bf16* p = WiL + ((t + 8) * 16 + ncol) * 136 + s * 32 + lhalf * 16;
        bfrag.q[0] = *(const uint4*)p;
        bfrag.q[1] = *(const uint4*)(p + 8);
        aZ = wmma_bf(af[s].v, bfrag.v, aZ);
      }
#pragma unroll
      for (int s = 0; s < 4; ++s) {
        Frag16 bfrag;
        const __bf16* p = WiL + ((t + 16) * 16 + ncol) * 136 + s * 32 + lhalf * 16;
        bfrag.q[0] = *(const uint4*)p;
        bfrag.q[1] = *(const uint4*)(p + 8);
        aN = wmma_bf(af[s].v, bfrag.v, aN);
      }
      const int colg = t * 16 + ncol;          // c index in [0,128)
      const float crb = biL[colg] + bhL[colg];
      const float czb = biL[GHD + colg] + bhL[GHD + colg];
      const float cnb = biL[2 * GHD + colg];
      const float bnv = bhL[2 * GHD + colg];
      const float wg0 = WgL[colg], wg1 = WgL[GHD + colg];
#pragma unroll
      for (int e = 0; e < 8; ++e) {
        float r  = fast_sigmoid(aR[e] + crb);
        float z  = fast_sigmoid(aZ[e] + czb);
        float nn = fast_tanh(aN[e] + cnb + r * bnv);
        float go = (1.f - z) * nn;
        pl0[e] = fmaf(go, wg0, pl0[e]);
        pl1[e] = fmaf(go, wg1, pl1[e]);
      }
    }
    // reduce logit partials across the 16 columns of each lane-half
#pragma unroll
    for (int m = 1; m <= 8; m <<= 1) {
#pragma unroll
      for (int e = 0; e < 8; ++e) {
        pl0[e] += __shfl_xor(pl0[e], m, 32);
        pl1[e] += __shfl_xor(pl1[e], m, 32);
      }
    }
    if (ncol == 0) {
      const int rbase = mbase + lhalf * 8;     // lanes 0 / 16 cover rows 0-7 / 8-15
#pragma unroll
      for (int e = 0; e < 8; ++e) {
        const int b = rbase + e;
        const float* gp = gumbel + (((size_t)b * NAG + i) * NAG + j) * 2;
        const float l0 = pl0[e] + bgL[0] + gp[0];
        const float l1 = pl1[e] + bgL[1] + gp[1];
        // argmax tie-break picks index 0 -> strict '>' for class 1; diagonal zeroed
        wout[((size_t)b * NAG + i) * NAG + j] = (i != j && l1 > l0) ? 1.f : 0.f;
      }
    }
  }
}

// ---------------- Kernel 3: gated aggregation + decode (fp32) ----------------
__global__ __launch_bounds__(256) void decode_kernel(
    const float* __restrict__ h, const float* __restrict__ w,
    const float* __restrict__ Wd, const float* __restrict__ bd,
    float* __restrict__ q) {
  __shared__ float WdL[ACT * 2 * HID];
  __shared__ float bdL[ACT];
  const int i = blockIdx.y;
  const int tid = threadIdx.x;
  for (int e = tid; e < ACT * 2 * HID; e += 256) WdL[e] = Wd[(size_t)i * ACT * 2 * HID + e];
  if (tid < ACT) bdL[tid] = bd[i * ACT + tid];
  __syncthreads();
  const int b = blockIdx.x * 256 + tid;
  float other[HID];
#pragma unroll
  for (int c = 0; c < HID; ++c) other[c] = 0.f;
  const float* wrow = w + ((size_t)b * NAG + i) * NAG;
  for (int jj = 0; jj < NAG; ++jj) {
    const float wv = wrow[jj];
    if (wv != 0.f) {
      const float* hj = h + ((size_t)b * NAG + jj) * HID;
#pragma unroll
      for (int c = 0; c < HID; ++c) other[c] = fmaf(wv, hj[c], other[c]);
    }
  }
  float qa[ACT];
#pragma unroll
  for (int a = 0; a < ACT; ++a) qa[a] = bdL[a];
  const float* hi = h + ((size_t)b * NAG + i) * HID;
  for (int c = 0; c < HID; ++c) {
    const float v = hi[c];
#pragma unroll
    for (int a = 0; a < ACT; ++a) qa[a] = fmaf(v, WdL[a * 2 * HID + c], qa[a]);
  }
  for (int c = 0; c < HID; ++c) {
    const float v = other[c];
#pragma unroll
    for (int a = 0; a < ACT; ++a) qa[a] = fmaf(v, WdL[a * 2 * HID + HID + c], qa[a]);
  }
  float* qp = q + ((size_t)b * NAG + i) * ACT;
#pragma unroll
  for (int a = 0; a < ACT; ++a) qp[a] = qa[a];
}

extern "C" void kernel_launch(void* const* d_in, const int* in_sizes, int n_in,
                              void* d_out, int out_size, void* d_ws, size_t ws_size,
                              hipStream_t stream) {
  (void)in_sizes; (void)n_in; (void)out_size; (void)ws_size;
  const float* obs    = (const float*)d_in[0];
  const float* gumbel = (const float*)d_in[1];
  const float* W1     = (const float*)d_in[2];
  const float* b1     = (const float*)d_in[3];
  const float* W2     = (const float*)d_in[4];
  const float* b2     = (const float*)d_in[5];
  const float* Wi     = (const float*)d_in[6];
  const float* bi     = (const float*)d_in[7];
  const float* bh     = (const float*)d_in[8];
  const float* Wg     = (const float*)d_in[9];
  const float* bg     = (const float*)d_in[10];
  const float* Wd     = (const float*)d_in[11];
  const float* bd     = (const float*)d_in[12];
  float* qout = (float*)d_out;

  char* ws = (char*)d_ws;
  float*  h   = (float*)ws;                                        // 8 MB
  __bf16* hbf = (__bf16*)(ws + (size_t)BATCH * NAG * HID * 4);     // 4 MB
  float*  w   = (float*)(ws + (size_t)BATCH * NAG * HID * 6);      // 1 MB

  encode_kernel<<<dim3(BATCH / 64, NAG), 64, 0, stream>>>(obs, W1, b1, W2, b2, h, hbf);
  pair_kernel<<<dim3(8, NAG * NAG), 256, 0, stream>>>(hbf, gumbel, Wi, bi, bh, Wg, bg, w);
  decode_kernel<<<dim3(BATCH / 256, NAG), 256, 0, stream>>>(h, w, Wd, bd, qout);
}
